// RNNModel_65420941852897
// MI455X (gfx1250) — matrix operand order repaired and checked by reading, not compiled, not run on hardware
//
#include <hip/hip_runtime.h>

// ---------------------------------------------------------------------------
// Block-diagonal LSTM LM for MI455X (gfx1250, wave32, WMMA + TDM).
//   T=128 B=32 D=1024 K=2 L=2 V=50257, DB=512, G=2048
// Strategy:
//   * all GEMMs via v_wmma_f32_16x16x32_bf16 (bf16 in, fp32 accum)
//   * decoder (422 GFLOP) dominates: per WG, 64-col Wdec strip staged in LDS
//     (bf16) once; A (x2) staged per 64-row M-block via the Tensor Data Mover
//     (tensor_load_to_lds + s_wait_tensorcnt) with TDM padding for a
//     bank-conflict-free LDS layout. Wdec streamed from HBM exactly once.
//   * recurrent scan: persistent kernel, 64 WGs x 8 waves = 512 waves =
//     512 gate tiles/step, agent-scope arrive/release grid barrier.
// ---------------------------------------------------------------------------

#define TT 128
#define BB 32
#define DD 1024
#define KK 2
#define LL 2
#define VV 50257
#define DBK 512      // D / K
#define GG 2048      // 4*D / K
#define FD 4096      // 4*D
#define OFF_H 205852672            // T*B*V
#define OFF_C (OFF_H + LL*BB*DD)   // + L*B*D
#define OFF_S (OFF_C + LL*BB*DD)   // scalar

#define LDSPITCH 1040              // padded row pitch (elements) in LDS
#define A_LDS_OFF (64 * LDSPITCH * 2)  // byte offset of A region (after B strip)

typedef __attribute__((ext_vector_type(16))) __bf16 bf16x16;
typedef __attribute__((ext_vector_type(8)))  float  floatx8;
typedef __attribute__((ext_vector_type(4)))  unsigned int u32x4;
typedef __attribute__((ext_vector_type(4)))  int  i32x4;
typedef __attribute__((ext_vector_type(8)))  int  i32x8;

__device__ __forceinline__ unsigned short f2bf(float f) {
  unsigned int u = __float_as_uint(f);
  unsigned int r = u + 0x7FFFu + ((u >> 16) & 1u);
  return (unsigned short)(r >> 16);
}

__device__ __forceinline__ float sigf(float x) {
  return 1.0f / (1.0f + __expf(-x));
}

// A fragment: 16x32 bf16 tile of a row-major (M x lda) matrix at (m0,k0).
// ISA layout: lanes 0-15 hold M=lane, K {0..7 -> e0..7, 16..23 -> e8..15};
// lanes 16-31 hold K {8..15, 24..31}.  => two contiguous 16B loads per lane.
__device__ __forceinline__ bf16x16 load_a_frag(const unsigned short* A, int lda,
                                               int m0, int k0, int lane) {
  const unsigned short* p = A + (size_t)(m0 + (lane & 15)) * lda + k0 + ((lane >> 4) << 3);
  union { u32x4 q[2]; bf16x16 v; } u;
  u.q[0] = *reinterpret_cast<const u32x4*>(p);
  u.q[1] = *reinterpret_cast<const u32x4*>(p + 16);
  return u.v;
}

// Same, but from the TDM-staged LDS A block: row pitch LDSPITCH elements with
// a 8-element pad inserted after element 512 of each row (TDM pad_interval).
__device__ __forceinline__ bf16x16 load_a_frag_lds(const unsigned short* Ablk,
                                                   int m0, int k0, int lane) {
  int row = m0 + (lane & 15);
  int ko  = k0 + ((lane >> 4) << 3);
  const unsigned short* p = Ablk + row * LDSPITCH + ko + ((ko >= 512) ? 8 : 0);
  union { u32x4 q[2]; bf16x16 v; } u;
  u.q[0] = *reinterpret_cast<const u32x4*>(p);
  u.q[1] = *reinterpret_cast<const u32x4*>(p + 16);
  return u.v;
}

// B fragment: 32x16 bf16 tile. Source stored TRANSPOSED as (N x ldk) row-major
// so each lane reads two 8-element bf16 chunks (two 16B loads).
// ISA layout: lane l holds column n0+(l&15), rows k0+8*(l>>4)+{0..7, 16..23}.
__device__ __forceinline__ bf16x16 load_b_frag(const unsigned short* Bt, int ldk,
                                               int n0, int k0, int lane) {
  const unsigned short* p = Bt + (size_t)(n0 + (lane & 15)) * ldk + k0 + ((lane >> 4) << 3);
  union { u32x4 q[2]; bf16x16 v; } u;
  u.q[0] = *reinterpret_cast<const u32x4*>(p);
  u.q[1] = *reinterpret_cast<const u32x4*>(p + 16);
  return u.v;
}

__device__ __forceinline__ floatx8 wmma_bf16(bf16x16 a, bf16x16 b, floatx8 c) {
  return __builtin_amdgcn_wmma_f32_16x16x32_bf16(false, a, false, b, (short)0, c,
                                                 false, false);
}

// ------------------------- grid-wide barrier --------------------------------
__device__ __forceinline__ void grid_sync(int* bar, int nblocks, int& sense) {
  __syncthreads();
  if (threadIdx.x == 0) {
    __threadfence();
    int prev = __hip_atomic_fetch_add(&bar[0], 1, __ATOMIC_ACQ_REL,
                                      __HIP_MEMORY_SCOPE_AGENT);
    if (prev == nblocks - 1) {
      __hip_atomic_store(&bar[0], 0, __ATOMIC_RELAXED, __HIP_MEMORY_SCOPE_AGENT);
      __hip_atomic_store(&bar[1], sense, __ATOMIC_RELEASE, __HIP_MEMORY_SCOPE_AGENT);
    } else {
      while (__hip_atomic_load(&bar[1], __ATOMIC_ACQUIRE,
                               __HIP_MEMORY_SCOPE_AGENT) != sense) {
        __builtin_amdgcn_s_sleep(1);
      }
    }
  }
  __syncthreads();
  __threadfence();
  sense ^= 1;
}

// ------------------------------ kernels -------------------------------------
__global__ void k_init(int* bar, float* out) {
  if (threadIdx.x == 0 && blockIdx.x == 0) {
    bar[0] = 0;
    bar[1] = 0;
    out[OFF_S] = 0.0f;  // the reference's trailing float32(0.0)
  }
}

// Wi/Wh (L,K,DB,G) fp32 -> transposed bf16 (L,K,G,DB)
__global__ void k_convert_w(const float* __restrict__ Wi, const float* __restrict__ Wh,
                            unsigned short* __restrict__ WiT,
                            unsigned short* __restrict__ WhT) {
  const long total = (long)LL * KK * GG * DBK;  // 4,194,304
  for (long o = (long)blockIdx.x * blockDim.x + threadIdx.x; o < total;
       o += (long)gridDim.x * blockDim.x) {
    long d  = o & (DBK - 1);
    long g  = (o >> 9) & (GG - 1);
    long lk = o >> 20;  // l*K + k
    long in = (lk * DBK + d) * GG + g;
    WiT[o] = f2bf(Wi[in]);
    WhT[o] = f2bf(Wh[in]);
  }
}

// x0b[tb*D + d] = bf16(emb[tokens[tb]*D + d])
__global__ void k_embed(const int* __restrict__ tokens, const float* __restrict__ emb,
                        unsigned short* __restrict__ x0b) {
  const int total = TT * BB * DD;
  for (int i = blockIdx.x * blockDim.x + threadIdx.x; i < total;
       i += gridDim.x * blockDim.x) {
    int tb = i >> 10, d = i & (DD - 1);
    x0b[i] = f2bf(emb[(size_t)tokens[tb] * DD + d]);
  }
}

// xg(4096 x 4096) = blockdiag(xin_b @ Wi) + bias.  One 16x16 tile per wave.
__global__ void k_xgates(const unsigned short* __restrict__ xinb,
                         const unsigned short* __restrict__ WiT_l,
                         const float* __restrict__ bias,
                         float* __restrict__ xg) {
  int tid = threadIdx.x, lane = tid & 31;
  int widx = blockIdx.x * 8 + (tid >> 5);       // 0..65535
  int mt = widx >> 8, nt = widx & 255;
  int m0 = mt * 16, n0 = nt * 16;
  int kblk = n0 >> 11, g0 = n0 & (GG - 1);
  const unsigned short* Bt = WiT_l + (size_t)kblk * GG * DBK;
  floatx8 acc = {};
#pragma unroll 4
  for (int kc = 0; kc < DBK; kc += 32) {
    bf16x16 a = load_a_frag(xinb, DD, m0, kblk * DBK + kc, lane);
    bf16x16 b = load_b_frag(Bt, DBK, g0, kc, lane);
    acc = wmma_bf16(a, b, acc);
  }
  int n = n0 + (lane & 15);
  int mb = m0 + ((lane >> 4) << 3);
  float bv = bias[n];
#pragma unroll
  for (int r = 0; r < 8; ++r) xg[(size_t)(mb + r) * FD + n] = acc[r] + bv;
}

// Persistent recurrent scan: 64 blocks x 256 threads = 512 waves = 512 tiles.
__global__ void __launch_bounds__(256)
k_scan(const float* __restrict__ xg, const unsigned short* __restrict__ WhT_l,
       const float* __restrict__ h0l, const float* __restrict__ c0l,
       float* __restrict__ cstate, unsigned short* __restrict__ hb,
       float* __restrict__ gates, unsigned short* __restrict__ xoutb,
       float* __restrict__ out, int layer, int sense0, int* bar) {
  const int NB = 64;
  int tid = threadIdx.x, lane = tid & 31;
  int widx = blockIdx.x * 8 + (tid >> 5);       // 0..511
  int nt = widx & 255, mt = widx >> 8;
  int n0 = nt * 16, m0 = mt * 16;
  int kblk = n0 >> 11, g0 = n0 & (GG - 1);
  const unsigned short* Bt = WhT_l + (size_t)kblk * GG * DBK;
  int n  = n0 + (lane & 15);
  int mb = m0 + ((lane >> 4) << 3);
  int gthread = blockIdx.x * 256 + tid;         // 0..16383
  int sense = sense0;

  for (int e = gthread; e < BB * DD; e += NB * 256) {
    cstate[e] = c0l[e];
    hb[e] = f2bf(h0l[e]);
  }
  grid_sync(bar, NB, sense);

  for (int t = 0; t < TT; ++t) {
    // ---- GEMM phase: gates tile = xg[t] tile + h @ Wh ----
    const float* xgt = xg + (size_t)t * (BB * FD);
    floatx8 acc;
#pragma unroll
    for (int r = 0; r < 8; ++r) acc[r] = xgt[(size_t)(mb + r) * FD + n];
#pragma unroll 4
    for (int kc = 0; kc < DBK; kc += 32) {
      bf16x16 a = load_a_frag(hb, DD, m0, kblk * DBK + kc, lane);
      bf16x16 b = load_b_frag(Bt, DBK, g0, kc, lane);
      acc = wmma_bf16(a, b, acc);
    }
#pragma unroll
    for (int r = 0; r < 8; ++r) gates[(size_t)(mb + r) * FD + n] = acc[r];
    grid_sync(bar, NB, sense);

    // ---- pointwise LSTM phase ----
    for (int e = gthread; e < BB * DD; e += NB * 256) {
      int b_ = e >> 10, d = e & (DD - 1);
      const float* gr = gates + (size_t)b_ * FD;
      float gi = gr[d], gf = gr[DD + d], gg = gr[2 * DD + d], go = gr[3 * DD + d];
      float cc = sigf(gf) * cstate[e] + sigf(gi) * tanhf(gg);
      float hh = sigf(go) * tanhf(cc);
      cstate[e] = cc;
      unsigned short hv = f2bf(hh);
      hb[e] = hv;
      xoutb[(size_t)t * (BB * DD) + e] = hv;
      if (t == TT - 1) {
        out[OFF_H + layer * BB * DD + e] = hh;
        out[OFF_C + layer * BB * DD + e] = cc;
      }
    }
    grid_sync(bar, NB, sense);
  }
}

// ---- TDM: stage one 64-row x 2048-byte A block into LDS at A_LDS_OFF with a
// 16B pad every 1024B (=> row pitch 2080B, bank-conflict-free ds_load_b128).
// D# per CDNA5 ISA (group0/group1 bit layout).
__device__ __forceinline__ void tdm_stage_a(const unsigned short* gsrc) {
#if __has_builtin(__builtin_amdgcn_tensor_load_to_lds)
  unsigned long long ga = (unsigned long long)(__UINTPTR_TYPE__)gsrc;
  u32x4 g0;
  g0[0] = 1u;                                   // count=1, is_restore=0
  g0[1] = (unsigned)A_LDS_OFF;                  // lds_addr (bytes)
  g0[2] = (unsigned)ga;                         // global_addr[31:0]
  g0[3] = (unsigned)(ga >> 32) | 0x80000000u;   // global_addr[56:32] | type=2
  i32x8 g1;
  g1[0] = 0x07D00000;   // data_size=1B, pad_enable, pad_interval=7(1024B), pad_amount=3(16B)
  g1[1] = 0x08000000;   // tensor_dim0 = 2048 (low16 -> bits 63:48)
  g1[2] = 0x00400000;   // tensor_dim0 hi=0 | tensor_dim1 = 64 (low16 -> bits 95:80)
  g1[3] = 0x08000000;   // tensor_dim1 hi=0 | tile_dim0 = 2048 (bits 127:112)
  g1[4] = 64;           // tile_dim1 = 64, tile_dim2 = 0
  g1[5] = 2048;         // tensor_dim0_stride lo32
  g1[6] = 0;            // tensor_dim0_stride hi | tensor_dim1_stride lo16
  g1[7] = 0;            // tensor_dim1_stride hi
  i32x4 z4 = {0, 0, 0, 0};
#if __clang_major__ >= 23
  i32x8 z8 = {0, 0, 0, 0, 0, 0, 0, 0};
  __builtin_amdgcn_tensor_load_to_lds(g0, g1, z4, z4, z8, 0);
#else
  __builtin_amdgcn_tensor_load_to_lds(g0, g1, z4, z4, 0);
#endif
  __builtin_amdgcn_s_wait_tensorcnt(0);
#endif
}

// Decoder: out(4096 x V) = x2 @ Wdec^T + bdec.
// Per WG: 64-col Wdec strip -> LDS (bf16, converted once, HBM read once);
// loop 64 M-blocks of 64 rows, A block TDM-staged into LDS; 8 waves =
// (4 col subtiles) x (2 row sub-blocks), 2 M-tiles per wave per B fragment.
__global__ void __launch_bounds__(256)
k_decode(const unsigned short* __restrict__ x2b, const float* __restrict__ Wdec,
         const float* __restrict__ bdec, float* __restrict__ out) {
  extern __shared__ unsigned short smem[];
  unsigned short* sB = smem;                        // 64 x LDSPITCH bf16
  unsigned short* sA = smem + 64 * LDSPITCH;        // 64 x LDSPITCH bf16

  int tid = threadIdx.x;
  int colbase = blockIdx.x * 64;

  for (int idx = tid; idx < 64 * DD; idx += 256) {
    int c_ = idx >> 10, kk = idx & (DD - 1);
    int gc = colbase + c_;
    float v = (gc < VV) ? Wdec[(size_t)gc * DD + kk] : 0.0f;
    sB[c_ * LDSPITCH + kk] = f2bf(v);
  }

  int w = tid >> 5, lane = tid & 31;
  int n0l = (w & 3) * 16;
  int sub = (w >> 2) * 32;
  int n = colbase + n0l + (lane & 15);
  float bias = (n < VV) ? bdec[n] : 0.0f;

  for (int mblk = 0; mblk < 64; ++mblk) {
    __syncthreads();  // B ready (first iter) / previous A block consumed
#if __has_builtin(__builtin_amdgcn_tensor_load_to_lds)
    if (tid < 32) tdm_stage_a(x2b + (size_t)mblk * 64 * DD);  // wave 0 issues DMA
#else
    {  // fallback: cooperative padded copy
      const unsigned short* gsrc = x2b + (size_t)mblk * 64 * DD;
      for (int idx = tid; idx < 64 * 128; idx += 256) {
        int r = idx >> 7, c8 = (idx & 127) * 8;
        u32x4 v = *reinterpret_cast<const u32x4*>(gsrc + r * DD + c8);
        *reinterpret_cast<u32x4*>(sA + r * LDSPITCH + c8 + ((c8 >= 512) ? 8 : 0)) = v;
      }
    }
#endif
    __syncthreads();  // A block staged

    floatx8 a0 = {}, a1 = {};
    for (int kc = 0; kc < DD; kc += 32) {
      bf16x16 bfr = load_b_frag(sB, LDSPITCH, n0l, kc, lane);   // ds_load_b128
      a0 = wmma_bf16(load_a_frag_lds(sA, sub,      kc, lane), bfr, a0);
      a1 = wmma_bf16(load_a_frag_lds(sA, sub + 16, kc, lane), bfr, a1);
    }
    if (n < VV) {
      int mbase = mblk * 64 + sub + ((lane >> 4) << 3);
#pragma unroll
      for (int r = 0; r < 8; ++r) {
        out[(size_t)(mbase + r)      * VV + n] = a0[r] + bias;
        out[(size_t)(mbase + 16 + r) * VV + n] = a1[r] + bias;
      }
    }
  }
}

// ------------------------------ launcher ------------------------------------
extern "C" void kernel_launch(void* const* d_in, const int* in_sizes, int n_in,
                              void* d_out, int out_size, void* d_ws, size_t ws_size,
                              hipStream_t stream) {
  const int*   tokens = (const int*)d_in[0];
  const float* h0     = (const float*)d_in[1];
  const float* c0     = (const float*)d_in[2];
  const float* emb    = (const float*)d_in[3];
  const float* Wi     = (const float*)d_in[4];
  const float* Wh     = (const float*)d_in[5];
  const float* bias   = (const float*)d_in[6];
  const float* Wdec   = (const float*)d_in[7];
  const float* bdec   = (const float*)d_in[8];
  float* out = (float*)d_out;

  // bump allocator over d_ws (total ~110 MB)
  char* ws = (char*)d_ws;
  size_t off = 0;
  auto alloc = [&](size_t bytes) -> char* {
    char* p = ws + off;
    off = (off + bytes + 255) & ~(size_t)255;
    return p;
  };
  const size_t WELEM = (size_t)LL * KK * GG * DBK;   // 4,194,304
  const size_t XELEM = (size_t)TT * BB * DD;         // 4,194,304
  int* bar                  = (int*)alloc(256);
  unsigned short* WiT       = (unsigned short*)alloc(WELEM * 2);
  unsigned short* WhT       = (unsigned short*)alloc(WELEM * 2);
  unsigned short* x0b       = (unsigned short*)alloc(XELEM * 2);
  unsigned short* x1b       = (unsigned short*)alloc(XELEM * 2);
  unsigned short* x2b       = (unsigned short*)alloc(XELEM * 2);
  float* xg                 = (float*)alloc((size_t)TT * BB * FD * 4);
  float* cstate             = (float*)alloc((size_t)BB * DD * 4);
  unsigned short* hb        = (unsigned short*)alloc((size_t)BB * DD * 2);
  float* gates              = (float*)alloc((size_t)BB * FD * 4);
  (void)ws_size; (void)in_sizes; (void)n_in; (void)out_size;

  k_init<<<1, 64, 0, stream>>>(bar, out);
  k_convert_w<<<2048, 256, 0, stream>>>(Wi, Wh, WiT, WhT);
  k_embed<<<2048, 256, 0, stream>>>(tokens, emb, x0b);

  // layer 0
  k_xgates<<<8192, 256, 0, stream>>>(x0b, WiT, bias, xg);
  k_scan<<<64, 256, 0, stream>>>(xg, WhT, h0, c0, cstate, hb, gates, x1b, out,
                                 0, /*sense0=*/1, bar);
  // layer 1 (barrier phase parity flips each k_scan: 257 barriers -> odd)
  k_xgates<<<8192, 256, 0, stream>>>(x1b, WiT + (size_t)KK * GG * DBK,
                                     bias + FD, xg);
  k_scan<<<64, 256, 0, stream>>>(xg, WhT + (size_t)KK * GG * DBK,
                                 h0 + BB * DD, c0 + BB * DD, cstate, hb, gates,
                                 x2b, out, 1, /*sense0=*/0, bar);

  // decoder: 786 WGs cover 786*64 = 50304 >= 50257 columns
  const size_t smem = (size_t)2 * 64 * LDSPITCH * sizeof(unsigned short);  // 260 KB
  k_decode<<<786, 256, smem, stream>>>(x2b, Wdec, bdec, out);
}
